// TransformerDecoder_75514114998741
// MI455X (gfx1250) — matrix operand order repaired and checked by reading, not compiled, not run on hardware
//
#include <hip/hip_runtime.h>

// ---------------- problem constants (match reference) ----------------
#define BB 2
#define TT 1024
#define VV 32000
#define EE 768
#define HH 12
#define LL 8
#define HS 64
#define FF_ 3072
#define MM (BB*TT)          // 2048 token rows

typedef __bf16  v16bf __attribute__((ext_vector_type(16)));
typedef __bf16  v8bf  __attribute__((ext_vector_type(8)));
typedef float   v8f   __attribute__((ext_vector_type(8)));

// ---------------------------------------------------------------------
// f32 -> bf16 conversion (grid-stride)
// ---------------------------------------------------------------------
__global__ __launch_bounds__(256)
void f32_to_bf16_kernel(const float* __restrict__ in, __bf16* __restrict__ out, long n) {
    long i = (long)blockIdx.x * blockDim.x + threadIdx.x;
    long stride = (long)gridDim.x * blockDim.x;
    for (; i < n; i += stride) out[i] = (__bf16)in[i];
}

// ---------------------------------------------------------------------
// x = tok_emb[idx] + pos_emb   (one block per token row)
// ---------------------------------------------------------------------
__global__ __launch_bounds__(256)
void embed_kernel(const int* __restrict__ idx, const float* __restrict__ tok,
                  const float* __restrict__ pos, float* __restrict__ x) {
    int bt = blockIdx.x;             // 0..B*T-1
    int t  = bt % TT;
    int token = idx[bt];
    const float* te = tok + (long)token * EE;
    const float* pe = pos + (long)t * EE;
    float* xr = x + (long)bt * EE;
    for (int e = threadIdx.x; e < EE; e += 256) xr[e] = te[e] + pe[e];
}

// ---------------------------------------------------------------------
// LayerNorm (row of E=768) -> bf16 output
// ---------------------------------------------------------------------
__global__ __launch_bounds__(256)
void layernorm_bf16_kernel(const float* __restrict__ x, const float* __restrict__ g,
                           const float* __restrict__ b, __bf16* __restrict__ out) {
    __shared__ float red[256];
    int row = blockIdx.x;
    const float* xr = x + (long)row * EE;
    float s = 0.f, s2 = 0.f;
    for (int e = threadIdx.x; e < EE; e += 256) { float v = xr[e]; s += v; s2 += v * v; }
    red[threadIdx.x] = s; __syncthreads();
    for (int o = 128; o > 0; o >>= 1) { if (threadIdx.x < o) red[threadIdx.x] += red[threadIdx.x + o]; __syncthreads(); }
    float mean = red[0] / EE; __syncthreads();
    red[threadIdx.x] = s2; __syncthreads();
    for (int o = 128; o > 0; o >>= 1) { if (threadIdx.x < o) red[threadIdx.x] += red[threadIdx.x + o]; __syncthreads(); }
    float var = red[0] / EE - mean * mean;
    float inv = rsqrtf(var + 1e-5f);
    for (int e = threadIdx.x; e < EE; e += 256) {
        float v = (xr[e] - mean) * inv * g[e] + b[e];
        out[(long)row * EE + e] = (__bf16)v;
    }
}

// ---------------------------------------------------------------------
// causal softmax over S rows; P = softmax(S*scale) (bf16), zero above diag
// row index = (b*H + h)*T + t
// ---------------------------------------------------------------------
__global__ __launch_bounds__(256)
void softmax_causal_kernel(const float* __restrict__ S, __bf16* __restrict__ P, float scale) {
    __shared__ float red[256];
    int row = blockIdx.x;
    int t = row % TT;
    const float* sr = S + (long)row * TT;
    __bf16* pr = P + (long)row * TT;

    float m = -1e30f;
    for (int c = threadIdx.x; c <= t; c += 256) m = fmaxf(m, sr[c] * scale);
    red[threadIdx.x] = m; __syncthreads();
    for (int o = 128; o > 0; o >>= 1) { if (threadIdx.x < o) red[threadIdx.x] = fmaxf(red[threadIdx.x], red[threadIdx.x + o]); __syncthreads(); }
    float rowmax = red[0]; __syncthreads();

    float sum = 0.f;
    for (int c = threadIdx.x; c <= t; c += 256) sum += __expf(sr[c] * scale - rowmax);
    red[threadIdx.x] = sum; __syncthreads();
    for (int o = 128; o > 0; o >>= 1) { if (threadIdx.x < o) red[threadIdx.x] += red[threadIdx.x + o]; __syncthreads(); }
    float inv = 1.f / red[0];

    for (int c = threadIdx.x; c < TT; c += 256) {
        float v = (c <= t) ? __expf(sr[c] * scale - rowmax) * inv : 0.f;
        pr[c] = (__bf16)v;
    }
}

// ---------------------------------------------------------------------
// LDS-staged WMMA bf16 GEMM:  C[z] = A[z](MxK) * op(B[z]) (+bias)(+ReLU)(+res)
//   TB=true : B is [N,K] row-major (C = A * B^T); B-chunk staged into LDS via
//             double-buffered GLOBAL_LOAD_ASYNC_TO_LDS_B128 (ASYNCcnt pipeline)
//   TB=false: B is [K,N] row-major (C = A * B); synchronous scatter-transpose
//   z decomposed as (zb, zh) = (z / zH, z % zH) with per-batch/per-head strides
//
// Block = 256 threads = 8 waves.
//   WN=2: block tile 128(M) x 128(N); wave tile 32x64 (waves 4x2)
//   WN=1: block tile 256(M) x  64(N); wave tile 32x64 (waves 8x1)
// K stepped by 32. B rows padded to 40 halves in LDS. Per K-step: preload all
// four B fragments (clause'd ds_load_b128s), then an uninterrupted 8-WMMA burst.
// ---------------------------------------------------------------------
template<int WN, bool TB, bool BIAS, bool RELU, bool RES, bool OUTBF>
__global__ __launch_bounds__(256)
void gemm_wmma_lds(const __bf16* __restrict__ A, long sAb, long sAh, int lda,
                   const __bf16* __restrict__ B, long sBb, long sBh, int ldb,
                   void* __restrict__ C, long sCb, long sCh, int ldc,
                   const float* __restrict__ bias, const float* __restrict__ Res,
                   int M, int N, int K, int zH) {
    constexpr int NBLK   = WN * 64;            // 128 or 64
    constexpr int MBLK   = 256 / WN;           // 128 or 256
    constexpr int LDSTR  = 40;                 // padded row stride (halves)
    constexpr int BUFH   = NBLK * LDSTR;       // halves per buffer
    __shared__ __bf16 ldsB[2 * BUFH];          // double-buffered B chunk

    const int tid  = threadIdx.x;
    const int wave = tid >> 5;
    const int lane = tid & 31;
    const int wn   = (WN == 2) ? (wave & 1) : 0;
    const int wm   = (WN == 2) ? (wave >> 1) : wave;

    const int z  = blockIdx.z;
    const int zb = z / zH;
    const int zh = z % zH;
    const __bf16* Ap = A + (long)zb * sAb + (long)zh * sAh;
    const __bf16* Bp = B + (long)zb * sBb + (long)zh * sBh;

    const int m0    = blockIdx.x * MBLK + wm * 32;   // wave owns rows m0..m0+31
    const int n0blk = blockIdx.y * NBLK;
    const int nw    = n0blk + wn * 64;               // wave's 64-col strip

    const int row  = lane & 15;          // A row within 16-tile
    const int kgA  = (lane >> 4) << 4;   // 0 or 16 (A-fragment K group)
    const int colB = lane & 15;          // B/C column within 16-tile
    const int kgB  = (lane >> 4) << 4;   // 0 or 16 (B-fragment K group)

    const v8f vz = {0.f,0.f,0.f,0.f,0.f,0.f,0.f,0.f};
    v8f acc[2][4] = {{vz,vz,vz,vz},{vz,vz,vz,vz}};

    const __bf16* arow0 = Ap + (long)(m0 +  0 + row) * lda;
    const __bf16* arow1 = Ap + (long)(m0 + 16 + row) * lda;

    // LDS byte address of buffer 0 (aperture bits live in addr[63:32])
    const unsigned ldsbase = (unsigned)(uintptr_t)(void*)&ldsB[0];
    // this wave's fragment base within a buffer (first of 4 cols strips)
    const __bf16* fragbase = &ldsB[(wn * 64 + colB) * LDSTR + kgB];

    // ---- stage one 32-deep B K-chunk into buffer `buf` ----
    auto stage = [&](int k0, int buf) {
        if (TB) {
            // async copy straight to LDS: b128 units (8 halves), row-contiguous
            constexpr int UNITS = NBLK * 4;
            #pragma unroll
            for (int u = tid; u < UNITS; u += 256) {
                const int r   = u >> 2;
                const int off = (u & 3) * 8;
                const __bf16* src = Bp + (long)(n0blk + r) * ldb + k0 + off;
                const unsigned dst = ldsbase + (unsigned)(buf * BUFH + r * LDSTR + off) * 2u;
                asm volatile("global_load_async_to_lds_b128 %0, %1, off"
                             :: "v"(dst), "v"((unsigned long long)(uintptr_t)src)
                             : "memory");
            }
        } else {
            // B[K,N]: read coalesced along N, scatter-transpose into LDS
            constexpr int UNITS = NBLK * 2;          // 16-half units
            for (int u = tid; u < UNITS; u += 256) {
                const int kk = u / (NBLK / 16);
                const int nc = (u % (NBLK / 16)) * 16;
                v16bf vsrc = *(const v16bf*)(Bp + (long)(k0 + kk) * ldb + n0blk + nc);
                #pragma unroll
                for (int i = 0; i < 16; ++i)
                    ldsB[buf * BUFH + (nc + i) * LDSTR + kk] = vsrc[i];
            }
        }
    };

    const int nk = K / 32;
    stage(0, 0);

    for (int i = 0; i < nk; ++i) {
        const int cur = i & 1;
        const int k0  = i * 32;
        if (TB) asm volatile("s_wait_asynccnt 0x0" ::: "memory");  // my buf[cur] staged
        __syncthreads();                       // all waves staged; buf[1-cur] free
        if (i + 1 < nk) stage(k0 + 32, 1 - cur);   // overlap next chunk with compute

        // A fragments (two 16-row tiles), direct from global
        v8bf lo0 = *(const v8bf*)(arow0 + k0 + kgA);
        v8bf hi0 = *(const v8bf*)(arow0 + k0 + kgA + 8);
        v16bf a0 = __builtin_shufflevector(lo0, hi0, 0,1,2,3,4,5,6,7,8,9,10,11,12,13,14,15);
        v8bf lo1 = *(const v8bf*)(arow1 + k0 + kgA);
        v8bf hi1 = *(const v8bf*)(arow1 + k0 + kgA + 8);
        v16bf a1 = __builtin_shufflevector(lo1, hi1, 0,1,2,3,4,5,6,7,8,9,10,11,12,13,14,15);
        if (k0 + 32 < K) {
            __builtin_prefetch(arow0 + k0 + 32, 0, 1);   // global_prefetch_b8
            __builtin_prefetch(arow1 + k0 + 32, 0, 1);
        }

        // preload all four B fragments (clause-able ds_load_b128 pairs) ...
        const __bf16* fb = fragbase + cur * BUFH;
        v16bf bfrag[4];
        #pragma unroll
        for (int t = 0; t < 4; ++t)
            bfrag[t] = *(const v16bf*)(fb + t * 16 * LDSTR);

        // ... then an uninterrupted 8-WMMA burst
        #pragma unroll
        for (int t = 0; t < 4; ++t) {
            acc[0][t] = __builtin_amdgcn_wmma_f32_16x16x32_bf16(
                            false, a0, false, bfrag[t], (short)0, acc[0][t], false, false);
            acc[1][t] = __builtin_amdgcn_wmma_f32_16x16x32_bf16(
                            false, a1, false, bfrag[t], (short)0, acc[1][t], false, false);
        }
    }

    // C/D layout: VGPR r -> M = r (lanes 0-15) or 8+r (lanes 16-31); N = lane&15
    const int rbase = (lane >> 4) * 8;
    #pragma unroll
    for (int wmi = 0; wmi < 2; ++wmi) {
        #pragma unroll
        for (int t = 0; t < 4; ++t) {
            const int cc = nw + t * 16 + colB;
            float bv = BIAS ? bias[cc] : 0.f;
            #pragma unroll
            for (int r = 0; r < 8; ++r) {
                const int rr = m0 + wmi * 16 + rbase + r;
                float vv = acc[wmi][t][r] + bv;
                if (RELU) vv = vv > 0.f ? vv : 0.f;
                const long cidx = (long)zb * sCb + (long)zh * sCh + (long)rr * ldc + cc;
                if (RES) vv += Res[cidx];
                if (OUTBF) ((__bf16*)C)[cidx] = (__bf16)vv;
                else       ((float*)C)[cidx]  = vv;
            }
        }
    }
}

// ---------------------------------------------------------------------
// host orchestration
// ---------------------------------------------------------------------
static inline void conv(const float* src, __bf16* dst, long n, hipStream_t s) {
    f32_to_bf16_kernel<<<2048, 256, 0, s>>>(src, dst, n);
}

extern "C" void kernel_launch(void* const* d_in, const int* in_sizes, int n_in,
                              void* d_out, int out_size, void* d_ws, size_t ws_size,
                              hipStream_t stream) {
    (void)in_sizes; (void)n_in; (void)out_size; (void)ws_size;

    const int*   idx     = (const int*)  d_in[0];
    const float* tok_emb = (const float*)d_in[1];
    const float* pos_emb = (const float*)d_in[2];
    const float* wq      = (const float*)d_in[3];
    const float* wk      = (const float*)d_in[4];
    const float* wv      = (const float*)d_in[5];
    const float* proj_w  = (const float*)d_in[6];
    const float* proj_b  = (const float*)d_in[7];
    const float* ln1_g   = (const float*)d_in[8];
    const float* ln1_b   = (const float*)d_in[9];
    const float* ln2_g   = (const float*)d_in[10];
    const float* ln2_b   = (const float*)d_in[11];
    const float* ff1_w   = (const float*)d_in[12];
    const float* ff1_b   = (const float*)d_in[13];
    const float* ff2_w   = (const float*)d_in[14];
    const float* ff2_b   = (const float*)d_in[15];
    const float* lnf_g   = (const float*)d_in[16];
    const float* lnf_b   = (const float*)d_in[17];
    const float* fc_w    = (const float*)d_in[18];
    const float* fc_b    = (const float*)d_in[19];
    float* logits = (float*)d_out;

    // ---- workspace carve-up (256B aligned) ----
    char* wsp = (char*)d_ws;
    auto carve = [&](size_t bytes) { char* p = wsp; wsp += (bytes + 255) & ~(size_t)255; return p; };
    float*  x     = (float*) carve((size_t)MM * EE * 4);          // residual stream
    __bf16* hbf   = (__bf16*)carve((size_t)MM * EE * 2);          // LN output (bf16)
    __bf16* qbf   = (__bf16*)carve((size_t)MM * EE * 2);
    __bf16* kbf   = (__bf16*)carve((size_t)MM * EE * 2);
    __bf16* vbf   = (__bf16*)carve((size_t)MM * EE * 2);
    __bf16* obf   = (__bf16*)carve((size_t)MM * EE * 2);          // attn output (concat heads)
    float*  Sbuf  = (float*) carve((size_t)BB * HH * TT * TT * 4);// attention scores
    __bf16* Pbuf  = (__bf16*)carve((size_t)BB * HH * TT * TT * 2);// softmax probs
    __bf16* ffmid = (__bf16*)carve((size_t)MM * FF_ * 2);
    __bf16* wlq   = (__bf16*)carve((size_t)EE * EE * 2);          // per-layer bf16 weights
    __bf16* wlk   = (__bf16*)carve((size_t)EE * EE * 2);
    __bf16* wlv   = (__bf16*)carve((size_t)EE * EE * 2);
    __bf16* wlp   = (__bf16*)carve((size_t)EE * EE * 2);
    __bf16* wlf1  = (__bf16*)carve((size_t)FF_ * EE * 2);
    __bf16* wlf2  = (__bf16*)carve((size_t)EE * FF_ * 2);
    __bf16* fcwbf = (__bf16*)carve((size_t)VV * EE * 2);

    const float scale = 1.0f / sqrtf((float)EE);   // reference scales by n_embd

    // once: fc_w -> bf16; embedding
    conv(fc_w, fcwbf, (long)VV * EE, stream);
    embed_kernel<<<MM, 256, 0, stream>>>(idx, tok_emb, pos_emb, x);

    for (int l = 0; l < LL; ++l) {
        // convert this layer's weights to bf16
        conv(wq     + (size_t)l * EE * EE,  wlq,  (long)EE * EE,  stream);
        conv(wk     + (size_t)l * EE * EE,  wlk,  (long)EE * EE,  stream);
        conv(wv     + (size_t)l * EE * EE,  wlv,  (long)EE * EE,  stream);
        conv(proj_w + (size_t)l * EE * EE,  wlp,  (long)EE * EE,  stream);
        conv(ff1_w  + (size_t)l * FF_ * EE, wlf1, (long)FF_ * EE, stream);
        conv(ff2_w  + (size_t)l * EE * FF_, wlf2, (long)EE * FF_, stream);

        // LN1
        layernorm_bf16_kernel<<<MM, 256, 0, stream>>>(x, ln1_g + l*EE, ln1_b + l*EE, hbf);

        // Q/K/V = h @ W^T   (M=2048, N=768, K=768)
        gemm_wmma_lds<2,true,false,false,false,true><<<dim3(MM/128, EE/128, 1), 256, 0, stream>>>(
            hbf, 0, 0, EE, wlq, 0, 0, EE, qbf, 0, 0, EE, nullptr, nullptr, MM, EE, EE, 1);
        gemm_wmma_lds<2,true,false,false,false,true><<<dim3(MM/128, EE/128, 1), 256, 0, stream>>>(
            hbf, 0, 0, EE, wlk, 0, 0, EE, kbf, 0, 0, EE, nullptr, nullptr, MM, EE, EE, 1);
        gemm_wmma_lds<2,true,false,false,false,true><<<dim3(MM/128, EE/128, 1), 256, 0, stream>>>(
            hbf, 0, 0, EE, wlv, 0, 0, EE, vbf, 0, 0, EE, nullptr, nullptr, MM, EE, EE, 1);

        // S = Q K^T  per (b,h): M=N=T, K=HS; A/B strides: batch T*E, head HS
        gemm_wmma_lds<2,true,false,false,false,false><<<dim3(TT/128, TT/128, BB*HH), 256, 0, stream>>>(
            qbf, (long)TT*EE, HS, EE,
            kbf, (long)TT*EE, HS, EE,
            Sbuf, (long)HH*TT*TT, (long)TT*TT, TT,
            nullptr, nullptr, TT, TT, HS, HH);

        // causal softmax -> P (bf16)
        softmax_causal_kernel<<<BB*HH*TT, 256, 0, stream>>>(Sbuf, Pbuf, scale);

        // O = P V  per (b,h): M=T, N=HS, K=T; B not transposed ([K=T, N=HS], ldb=E)
        gemm_wmma_lds<1,false,false,false,false,true><<<dim3(TT/256, HS/64, BB*HH), 256, 0, stream>>>(
            Pbuf, (long)HH*TT*TT, (long)TT*TT, TT,
            vbf,  (long)TT*EE, HS, EE,
            obf,  (long)TT*EE, HS, EE,
            nullptr, nullptr, TT, HS, TT, HH);

        // x = x + O @ proj_w^T + proj_b   (f32 out, residual)
        gemm_wmma_lds<2,true,true,false,true,false><<<dim3(MM/128, EE/128, 1), 256, 0, stream>>>(
            obf, 0, 0, EE, wlp, 0, 0, EE, x, 0, 0, EE,
            proj_b + l*EE, x, MM, EE, EE, 1);

        // LN2
        layernorm_bf16_kernel<<<MM, 256, 0, stream>>>(x, ln2_g + l*EE, ln2_b + l*EE, hbf);

        // ffmid = relu(h @ ff1_w^T + ff1_b)   (M=2048, N=3072, K=768)
        gemm_wmma_lds<2,true,true,true,false,true><<<dim3(MM/128, FF_/128, 1), 256, 0, stream>>>(
            hbf, 0, 0, EE, wlf1, 0, 0, EE, ffmid, 0, 0, FF_,
            ff1_b + l*FF_, nullptr, MM, FF_, EE, 1);

        // x = x + ffmid @ ff2_w^T + ff2_b     (M=2048, N=768, K=3072)
        gemm_wmma_lds<2,true,true,false,true,false><<<dim3(MM/128, EE/128, 1), 256, 0, stream>>>(
            ffmid, 0, 0, FF_, wlf2, 0, 0, FF_, x, 0, 0, EE,
            ff2_b + l*EE, x, MM, EE, FF_, 1);
    }

    // final LN + logits = h @ fc_w^T + fc_b   (M=2048, N=32000, K=768)
    layernorm_bf16_kernel<<<MM, 256, 0, stream>>>(x, lnf_g, lnf_b, hbf);
    gemm_wmma_lds<2,true,true,false,false,false><<<dim3(MM/128, VV/128, 1), 256, 0, stream>>>(
        hbf, 0, 0, EE, fcwbf, 0, 0, EE, logits, 0, 0, VV,
        fc_b, nullptr, MM, VV, EE, 1);
}
